// CELPNet_18305150616355
// MI455X (gfx1250) — matrix operand order repaired and checked by reading, not compiled, not run on hardware
//
#include <hip/hip_runtime.h>
#include <hip/hip_bf16.h>
#include <math.h>

// ---------------------------------------------------------------------------
// CELPNet forward for MI455X (gfx1250, wave32, WMMA bf16 16x16x32).
// B=512, T=204, FEAT=20, COND=256, SUB=40, steps = nb_frames*4 (=800).
//
//  * Weights packed once into WMMA-B-fragment-native bf16 layout; GRU weights
//    gate-interleaved so (jt,kt) -> 3 contiguous gate fragments (3KB).
//  * Cond net: WMMA GEMM+tanh with n-tile-pair fusion (1 A-frag -> 2 WMMAs),
//    convs are plain GEMMs with K=768 ([B,T,C] row-major = implicit im2col).
//  * Scan: ONE persistent kernel, 16 WGs x 32 batch rows, 800 steps in-kernel,
//    GRU states ping-ponged in LDS, m-tile-pair fusion so each WG streams the
//    2.7MB (L2-resident) weight set exactly once per step.
//  * Gate math uses gfx1250 V_TANH_F32 (single trans op) when available.
// ---------------------------------------------------------------------------

typedef __attribute__((ext_vector_type(16))) __bf16 bf16x16;
typedef __attribute__((ext_vector_type(8)))  float  f32x8;
typedef __attribute__((ext_vector_type(4)))  unsigned int u32x4;

union FragU { bf16x16 v; u32x4 q[2]; };

__device__ __forceinline__ f32x8 wmma_bf16(bf16x16 a, bf16x16 b, f32x8 c) {
  // v_wmma_f32_16x16x32_bf16  D = A(16x32) * B(32x16) + C
  return __builtin_amdgcn_wmma_f32_16x16x32_bf16(false, a, false, b, (short)0, c,
                                                 false, false);
}

// A-fragment (16x32 bf16, per-ISA layout):
//   lane<16 : M=lane,    K = {kb+0..7, kb+16..23}
//   lane>=16: M=lane-16, K = {kb+8..15, kb+24..31}
// rowPtr must already include the (lane<16 ? 0 : 8) K-offset.
__device__ __forceinline__ bf16x16 load_afrag(const __bf16* rowPtr) {
  FragU f;
  f.q[0] = *(const u32x4*)(rowPtr);
  f.q[1] = *(const u32x4*)(rowPtr + 16);
  return f.v;
}

// B-fragment: 32 contiguous bytes per lane (pointer already includes lane*16).
__device__ __forceinline__ bf16x16 ldfrag(const __bf16* p) {
  FragU f;
  f.q[0] = *(const u32x4*)(p);
  f.q[1] = *(const u32x4*)(p + 8);
  return f.v;
}

// gfx1250 has a hardware tanh transcendental (V_TANH_F32); use it when clang
// exposes the builtin, otherwise fall back to libm.
__device__ __forceinline__ float fast_tanh(float x) {
#if __has_builtin(__builtin_amdgcn_tanhf)
  return __builtin_amdgcn_tanhf(x);
#else
  return tanhf(x);
#endif
}
// sigmoid(x) = 0.5*tanh(x/2) + 0.5  (one trans op + fma instead of exp/div)
__device__ __forceinline__ float fast_sigm(float x) {
  return __builtin_fmaf(0.5f, fast_tanh(0.5f * x), 0.5f);
}

// ---------------------------------------------------------------------------
// Prep: pack weight [OUT][IN] (GEMM B = W^T, K=IN, N=OUT) into fragment layout
// with zero padding. Packed element: lane,e -> N = base_n + lane%16,
// K = kt*32 + (lane<16 ? 0 : 8) + (e<8 ? e : 8+e).
// mode 0: frag = nt*nKt + kt                       (plain [OUT][IN])
// mode 1: same frag order, conv src [CO][CI][3]:   src[n*768 + (k%256)*3 + k/256]
// mode 2: GRU gate-interleave: frag = (jt*nKt+kt)*3 + g,  n = g*256 + jt*16 + ...
// ---------------------------------------------------------------------------
__global__ void pack_weight(const float* __restrict__ src, __bf16* __restrict__ dst,
                            int OUT, int IN, int nKt, int mode, int total) {
  int idx = blockIdx.x * blockDim.x + threadIdx.x;
  if (idx >= total) return;
  int e    = idx & 15;
  int lane = (idx >> 4) & 31;
  int frag = idx >> 9;
  int n, kt;
  if (mode == 2) {
    int g = frag % 3, t = frag / 3;
    kt = t % nKt;
    int jt = t / nKt;
    n = g * 256 + jt * 16 + (lane & 15);
  } else {
    kt = frag % nKt;
    n = (frag / nKt) * 16 + (lane & 15);
  }
  int k = kt * 32 + ((lane < 16) ? 0 : 8) + ((e < 8) ? e : 8 + e);
  float v = 0.0f;
  if (n < OUT && k < IN) {
    if (mode == 1) {
      int ci = k & 255, tap = k >> 8;
      v = src[(long)n * 768 + ci * 3 + tap];
    } else {
      v = src[(long)n * IN + k];
    }
  }
  dst[idx] = (__bf16)v;
}

// features [512*204][20] fp32 -> bf16 [512*204][32] zero-padded
__global__ void pack_features(const float* __restrict__ f, __bf16* __restrict__ o,
                              int total) {
  int idx = blockIdx.x * blockDim.x + threadIdx.x;
  if (idx >= total) return;
  int c = idx & 31, r = idx >> 5;
  o[idx] = (__bf16)((c < 20) ? f[r * 20 + c] : 0.0f);
}

// ---------------------------------------------------------------------------
// Generic WMMA GEMM + tanh (cond net). One 16x32 output tile per wave
// (n-tile pair: one A-fragment feeds two WMMAs -> halves A traffic).
// B fragments are software-pipelined one k-tile ahead so the WMMA never waits
// on a just-issued load. A row r -> offset (r/Tout)*TinCin + (r%Tout)*Cin.
// ---------------------------------------------------------------------------
__global__ void __launch_bounds__(256)
gemm_tanh(const __bf16* __restrict__ A, const __bf16* __restrict__ Bp,
          const float* __restrict__ bias, __bf16* __restrict__ Out,
          int Mtiles, int nKt, int Tout, int TinCin, int Cin, int N) {
  int lane = threadIdx.x & 31;
  int wave = threadIdx.x >> 5;
  int tile = blockIdx.x * 8 + wave;            // indexes (mt, nt-pair)
  const int NP = 8;                            // N=256 -> 8 n-tile pairs
  if (tile >= Mtiles * NP) return;             // wave-uniform: EXEC stays full
  int mt = tile / NP, ntp = tile - mt * NP;

  int mrow = mt * 16 + (lane & 15);
  long aoff = (long)(mrow / Tout) * TinCin + (long)(mrow % Tout) * Cin;
  const __bf16* arow = A + aoff + ((lane < 16) ? 0 : 8);

  int ncol0 = ntp * 32 + (lane & 15);
  int ncol1 = ncol0 + 16;
  float bv0 = bias[ncol0], bv1 = bias[ncol1];
  f32x8 acc0, acc1;
#pragma unroll
  for (int i = 0; i < 8; ++i) { acc0[i] = bv0; acc1[i] = bv1; }

  const __bf16* bb0 = Bp + (long)(ntp * 2) * nKt * 512 + lane * 16;
  const __bf16* bb1 = bb0 + (long)nKt * 512;
  // 1-deep pipeline on B fragments (the +1 over-read on the last iteration
  // stays inside the packed-weights region of the workspace and is unused).
  bf16x16 b0c = ldfrag(bb0);
  bf16x16 b1c = ldfrag(bb1);
  for (int kt = 0; kt < nKt; ++kt) {
    bf16x16 a = load_afrag(arow + kt * 32);
    bf16x16 b0n = ldfrag(bb0 + (kt + 1) * 512);
    bf16x16 b1n = ldfrag(bb1 + (kt + 1) * 512);
    acc0 = wmma_bf16(a, b0c, acc0);
    acc1 = wmma_bf16(a, b1c, acc1);
    b0c = b0n;
    b1c = b1n;
  }
  long orow = (long)(mt * 16 + ((lane < 16) ? 0 : 8));
#pragma unroll
  for (int r = 0; r < 8; ++r) {
    Out[(orow + r) * N + ncol0] = (__bf16)fast_tanh(acc0[r]);
    Out[(orow + r) * N + ncol1] = (__bf16)fast_tanh(acc1[r]);
  }
}

// ---------------------------------------------------------------------------
// Persistent autoregressive scan. 16 blocks x 256 threads (8 waves), 32 batch
// rows per block. Dynamic LDS partition (bf16 elems):
//   xcat[32][320] | tact[32][256] | tact2[32][256] | h[3][2][32][256] | outb[32][48]
// = 77312 elems = 151 KB  (gfx1250: up to 320 KB per workgroup)
// The d1/d2/GRU GEMM jobs fuse both m-tiles (batch rows 0-15 and 16-31) so each B
// (weight) fragment is loaded once per workgroup per step.
// ---------------------------------------------------------------------------
#define SMEM_ELEMS (32 * 320 + 2 * 32 * 256 + 3 * 2 * 32 * 256 + 32 * 48)

__global__ void __launch_bounds__(256, 1)
scan_kernel(const __bf16* __restrict__ cond,
            const __bf16* wd1, const __bf16* wd2,
            const __bf16* wih1, const __bf16* whh1,
            const __bf16* wih2, const __bf16* whh2,
            const __bf16* wih3, const __bf16* whh3,
            const __bf16* wow,
            const float* __restrict__ d1b, const float* __restrict__ d2b,
            const float* bih1, const float* bhh1,
            const float* bih2, const float* bhh2,
            const float* bih3, const float* bhh3,
            const float* __restrict__ obias,
            float* __restrict__ out, int steps, int condT) {
  extern __shared__ __bf16 smem[];
  __bf16* s_xcat  = smem;                      // [32][320]
  __bf16* s_tact  = s_xcat + 32 * 320;         // [32][256]
  __bf16* s_tact2 = s_tact + 32 * 256;         // [32][256]
  __bf16* s_h     = s_tact2 + 32 * 256;        // [3][2][32][256]
  __bf16* s_outb  = s_h + 3 * 2 * 32 * 256;    // [32][48]

  const int tid  = threadIdx.x;
  const int lane = tid & 31;
  const int wave = tid >> 5;
  const int b0   = blockIdx.x * 32;
  const int kofs = (lane < 16) ? 0 : 8;
  const int mloc = lane & 15;

  const __bf16* wihA[3] = {wih1, wih2, wih3};
  const __bf16* whhA[3] = {whh1, whh2, whh3};
  const float*  bihA[3] = {bih1, bih2, bih3};
  const float*  bhhA[3] = {bhh1, bhh2, bhh3};

  // zero GRU states + prev output
  for (int i = tid; i < 3 * 2 * 32 * 256; i += 256) s_h[i] = (__bf16)0.0f;
  for (int i = tid; i < 32 * 48; i += 256) s_outb[i] = (__bf16)0.0f;
  __syncthreads();

  const long ostride = (long)steps * 40;

  for (int s = 0; s < steps; ++s) {
    const int cur = s & 1, nxt = cur ^ 1;
    const int frame = s >> 2;

    // ---- build xcat = [cond(256) | prev(40) | pad(24)] -------------------
    for (int idx = tid; idx < 32 * 320; idx += 256) {
      int i = idx / 320, c = idx - i * 320;
      __bf16 v = (__bf16)0.0f;
      if (c < 256)       v = cond[((long)(b0 + i) * condT + frame) * 256 + c];
      else if (c < 296)  v = s_outb[i * 48 + (c - 256)];
      s_xcat[idx] = v;
    }
    __syncthreads();

    // ---- d1: tanh(xcat @ d1_w^T + b)   K=320, N=256, both m-tiles --------
    for (int nt = wave; nt < 16; nt += 8) {
      int ncol = nt * 16 + mloc;
      float bv = d1b[ncol];
      f32x8 acc0, acc1;
#pragma unroll
      for (int i = 0; i < 8; ++i) { acc0[i] = bv; acc1[i] = bv; }
      const __bf16* ar0 = s_xcat + mloc * 320 + kofs;
      const __bf16* ar1 = s_xcat + (16 + mloc) * 320 + kofs;
      const __bf16* bb = wd1 + (long)nt * 10 * 512 + lane * 16;
      bf16x16 bc = ldfrag(bb);
      for (int kt = 0; kt < 10; ++kt) {
        bf16x16 bn = ldfrag(bb + (kt + 1) * 512);   // benign +1 over-read
        acc0 = wmma_bf16(load_afrag(ar0 + kt * 32), bc, acc0);
        acc1 = wmma_bf16(load_afrag(ar1 + kt * 32), bc, acc1);
        bc = bn;
      }
#pragma unroll
      for (int r = 0; r < 8; ++r) {
        s_tact[(kofs + r) * 256 + ncol]      = (__bf16)fast_tanh(acc0[r]);
        s_tact[(16 + kofs + r) * 256 + ncol] = (__bf16)fast_tanh(acc1[r]);
      }
    }
    __syncthreads();

    // ---- d2: tanh(t @ d2_w^T + b)   K=256, N=256, both m-tiles -----------
    for (int nt = wave; nt < 16; nt += 8) {
      int ncol = nt * 16 + mloc;
      float bv = d2b[ncol];
      f32x8 acc0, acc1;
#pragma unroll
      for (int i = 0; i < 8; ++i) { acc0[i] = bv; acc1[i] = bv; }
      const __bf16* ar0 = s_tact + mloc * 256 + kofs;
      const __bf16* ar1 = s_tact + (16 + mloc) * 256 + kofs;
      const __bf16* bb = wd2 + (long)nt * 8 * 512 + lane * 16;
      bf16x16 bc = ldfrag(bb);
      for (int kt = 0; kt < 8; ++kt) {
        bf16x16 bn = ldfrag(bb + (kt + 1) * 512);   // benign +1 over-read
        acc0 = wmma_bf16(load_afrag(ar0 + kt * 32), bc, acc0);
        acc1 = wmma_bf16(load_afrag(ar1 + kt * 32), bc, acc1);
        bc = bn;
      }
#pragma unroll
      for (int r = 0; r < 8; ++r) {
        s_tact2[(kofs + r) * 256 + ncol]      = (__bf16)fast_tanh(acc0[r]);
        s_tact2[(16 + kofs + r) * 256 + ncol] = (__bf16)fast_tanh(acc1[r]);
      }
    }
    __syncthreads();

    // ---- 3 stacked GRU cells: gates (r,z,n) x both m-tiles in registers --
#pragma unroll
    for (int l = 0; l < 3; ++l) {
      const __bf16* xb = (l == 0) ? s_tact2 : (s_h + ((l - 1) * 2 + nxt) * 8192);
      const __bf16* hb = s_h + (l * 2 + cur) * 8192;
      __bf16*       hn = s_h + (l * 2 + nxt) * 8192;
      const __bf16* wih = wihA[l];
      const __bf16* whh = whhA[l];
      const float*  bi  = bihA[l];
      const float*  bh  = bhhA[l];

      for (int jt = wave; jt < 16; jt += 8) {
        int jcol = jt * 16 + mloc;
        f32x8 gi0a, gi1a, gi2a, gh0a, gh1a, gh2a;
        f32x8 gi0b, gi1b, gi2b, gh0b, gh1b, gh2b;
        {
          float a0 = bi[jcol], a1 = bi[256 + jcol], a2 = bi[512 + jcol];
          float c0 = bh[jcol], c1 = bh[256 + jcol], c2 = bh[512 + jcol];
#pragma unroll
          for (int i = 0; i < 8; ++i) {
            gi0a[i] = a0; gi1a[i] = a1; gi2a[i] = a2;
            gh0a[i] = c0; gh1a[i] = c1; gh2a[i] = c2;
            gi0b[i] = a0; gi1b[i] = a1; gi2b[i] = a2;
            gh0b[i] = c0; gh1b[i] = c1; gh2b[i] = c2;
          }
        }
        const __bf16* xr0 = xb + mloc * 256 + kofs;
        const __bf16* xr1 = xb + (16 + mloc) * 256 + kofs;
        const __bf16* hr0 = hb + mloc * 256 + kofs;
        const __bf16* hr1 = hb + (16 + mloc) * 256 + kofs;
        // gate-interleaved frag order: ((jt*8 + kt)*3 + g)
        const __bf16* bbi = wih + ((long)jt * 8) * 3 * 512 + lane * 16;
        const __bf16* bbh = whh + ((long)jt * 8) * 3 * 512 + lane * 16;
        for (int kt = 0; kt < 8; ++kt) {
          bf16x16 ax0 = load_afrag(xr0 + kt * 32);
          bf16x16 ax1 = load_afrag(xr1 + kt * 32);
          bf16x16 ah0 = load_afrag(hr0 + kt * 32);
          bf16x16 ah1 = load_afrag(hr1 + kt * 32);
          const __bf16* pi = bbi + (long)kt * 3 * 512;
          const __bf16* ph = bbh + (long)kt * 3 * 512;
          bf16x16 w0 = ldfrag(pi);
          gi0a = wmma_bf16(ax0, w0, gi0a); gi0b = wmma_bf16(ax1, w0, gi0b);
          bf16x16 w1 = ldfrag(pi + 512);
          gi1a = wmma_bf16(ax0, w1, gi1a); gi1b = wmma_bf16(ax1, w1, gi1b);
          bf16x16 w2 = ldfrag(pi + 1024);
          gi2a = wmma_bf16(ax0, w2, gi2a); gi2b = wmma_bf16(ax1, w2, gi2b);
          bf16x16 v0 = ldfrag(ph);
          gh0a = wmma_bf16(ah0, v0, gh0a); gh0b = wmma_bf16(ah1, v0, gh0b);
          bf16x16 v1 = ldfrag(ph + 512);
          gh1a = wmma_bf16(ah0, v1, gh1a); gh1b = wmma_bf16(ah1, v1, gh1b);
          bf16x16 v2 = ldfrag(ph + 1024);
          gh2a = wmma_bf16(ah0, v2, gh2a); gh2b = wmma_bf16(ah1, v2, gh2b);
        }
#pragma unroll
        for (int r = 0; r < 8; ++r) {
          int row0 = kofs + r, row1 = 16 + kofs + r;
          float rg = fast_sigm(gi0a[r] + gh0a[r]);
          float zg = fast_sigm(gi1a[r] + gh1a[r]);
          float ng = fast_tanh(gi2a[r] + rg * gh2a[r]);
          float hv = (float)hb[row0 * 256 + jcol];
          hn[row0 * 256 + jcol] = (__bf16)((1.0f - zg) * ng + zg * hv);
          rg = fast_sigm(gi0b[r] + gh0b[r]);
          zg = fast_sigm(gi1b[r] + gh1b[r]);
          ng = fast_tanh(gi2b[r] + rg * gh2b[r]);
          hv = (float)hb[row1 * 256 + jcol];
          hn[row1 * 256 + jcol] = (__bf16)((1.0f - zg) * ng + zg * hv);
        }
      }
      __syncthreads();
    }

    // ---- output head: tanh(h3 @ ow^T + ob)  N=40 (pad 48) ----------------
    if (wave < 6) {                               // wave-uniform, EXEC full
      int mt = wave / 3, nt = wave - mt * 3;
      int ncol = nt * 16 + mloc;
      float bv = (ncol < 40) ? obias[ncol] : 0.0f;
      f32x8 acc;
#pragma unroll
      for (int i = 0; i < 8; ++i) acc[i] = bv;
      const __bf16* arow = s_h + (2 * 2 + nxt) * 8192 + (mt * 16 + mloc) * 256 + kofs;
      const __bf16* bb = wow + (long)nt * 8 * 512 + lane * 16;
      for (int kt = 0; kt < 8; ++kt) {
        bf16x16 a = load_afrag(arow + kt * 32);
        acc = wmma_bf16(a, ldfrag(bb + kt * 512), acc);
      }
      int br = mt * 16 + kofs;
#pragma unroll
      for (int r = 0; r < 8; ++r) {
        float v = fast_tanh(acc[r]);
        if (ncol < 40) {
          s_outb[(br + r) * 48 + ncol] = (__bf16)v;                 // next prev
          out[(long)(b0 + br + r) * ostride + (long)s * 40 + ncol] = v;
        }
      }
    }
    __syncthreads();
  }
}

// ---------------------------------------------------------------------------
extern "C" void kernel_launch(void* const* d_in, const int* in_sizes, int n_in,
                              void* d_out, int out_size, void* d_ws, size_t ws_size,
                              hipStream_t stream) {
  (void)in_sizes; (void)n_in; (void)ws_size;
  const float* features = (const float*)d_in[0];
  const float* fd1_w = (const float*)d_in[1];  const float* fd1_b = (const float*)d_in[2];
  const float* c1_w  = (const float*)d_in[3];  const float* c1_b  = (const float*)d_in[4];
  const float* c2_w  = (const float*)d_in[5];  const float* c2_b  = (const float*)d_in[6];
  const float* fd2_w = (const float*)d_in[7];  const float* fd2_b = (const float*)d_in[8];
  const float* d1_w  = (const float*)d_in[9];  const float* d1_b  = (const float*)d_in[10];
  const float* d2_w  = (const float*)d_in[11]; const float* d2_b  = (const float*)d_in[12];
  const float* g_wih[3] = {(const float*)d_in[13], (const float*)d_in[17], (const float*)d_in[21]};
  const float* g_whh[3] = {(const float*)d_in[14], (const float*)d_in[18], (const float*)d_in[22]};
  const float* g_bih[3] = {(const float*)d_in[15], (const float*)d_in[19], (const float*)d_in[23]};
  const float* g_bhh[3] = {(const float*)d_in[16], (const float*)d_in[20], (const float*)d_in[24]};
  const float* ow_w  = (const float*)d_in[25]; const float* ow_b  = (const float*)d_in[26];

  const int steps = out_size / (512 * 40);     // = nb_frames * 4

  // ---- workspace layout (~118 MB), 256B aligned chunks -------------------
  char* ws = (char*)d_ws;
  size_t off = 0;
  auto alloc = [&](size_t elems) -> __bf16* {
    __bf16* p = (__bf16*)(ws + off);
    off += ((elems * sizeof(__bf16) + 255) / 256) * 256;
    return p;
  };
  __bf16* wp_fd1 = alloc((size_t)16 * 1 * 512);
  __bf16* wp_c1  = alloc((size_t)16 * 24 * 512);
  __bf16* wp_c2  = alloc((size_t)16 * 24 * 512);
  __bf16* wp_fd2 = alloc((size_t)16 * 8 * 512);
  __bf16* wp_d1  = alloc((size_t)16 * 10 * 512);
  __bf16* wp_d2  = alloc((size_t)16 * 8 * 512);
  __bf16* wp_ih[3]; __bf16* wp_hh[3];
  for (int l = 0; l < 3; ++l) {
    wp_ih[l] = alloc((size_t)48 * 8 * 512);
    wp_hh[l] = alloc((size_t)48 * 8 * 512);
  }
  __bf16* wp_ow = alloc((size_t)3 * 8 * 512);
  __bf16* featb = alloc((size_t)512 * 204 * 32);
  __bf16* bufA  = alloc((size_t)512 * 204 * 256);
  __bf16* bufB  = alloc((size_t)512 * 204 * 256);

  // ---- prep: pack weights + features -------------------------------------
  auto packW = [&](const float* src, __bf16* dst, int OUT, int IN, int nKt,
                   int Ntiles, int mode) {
    int total = Ntiles * nKt * 512;
    pack_weight<<<(total + 255) / 256, 256, 0, stream>>>(src, dst, OUT, IN, nKt,
                                                         mode, total);
  };
  packW(fd1_w, wp_fd1, 256, 20, 1, 16, 0);
  packW(c1_w,  wp_c1,  256, 768, 24, 16, 1);
  packW(c2_w,  wp_c2,  256, 768, 24, 16, 1);
  packW(fd2_w, wp_fd2, 256, 256, 8, 16, 0);
  packW(d1_w,  wp_d1,  256, 296, 10, 16, 0);
  packW(d2_w,  wp_d2,  256, 256, 8, 16, 0);
  for (int l = 0; l < 3; ++l) {
    packW(g_wih[l], wp_ih[l], 768, 256, 8, 48, 2);   // gate-interleaved
    packW(g_whh[l], wp_hh[l], 768, 256, 8, 48, 2);   // gate-interleaved
  }
  packW(ow_w, wp_ow, 40, 256, 8, 3, 0);
  {
    int total = 512 * 204 * 32;
    pack_features<<<(total + 255) / 256, 256, 0, stream>>>(features, featb, total);
  }

  // ---- cond net: 4 WMMA GEMM+tanh (n-tile-pair fused) --------------------
  auto gemm = [&](const __bf16* A, const __bf16* Bp, const float* bias,
                  __bf16* Out, int Mtiles, int nKt, int Tout, int TinCin, int Cin) {
    int tiles = Mtiles * 8;                     // 8 n-tile pairs
    gemm_tanh<<<(tiles + 7) / 8, 256, 0, stream>>>(A, Bp, bias, Out, Mtiles,
                                                   nKt, Tout, TinCin, Cin, 256);
  };
  gemm(featb, wp_fd1, fd1_b, bufA, 512 * 204 / 16, 1, 204, 204 * 32, 32);    // fd1
  gemm(bufA,  wp_c1,  c1_b,  bufB, 512 * 202 / 16, 24, 202, 204 * 256, 256); // conv1
  gemm(bufB,  wp_c2,  c2_b,  bufA, 512 * 200 / 16, 24, 200, 202 * 256, 256); // conv2
  gemm(bufA,  wp_fd2, fd2_b, bufB, 512 * 200 / 16, 8, 200, 200 * 256, 256);  // fd2 -> cond

  // ---- persistent autoregressive scan ------------------------------------
  scan_kernel<<<16, 256, SMEM_ELEMS * sizeof(__bf16), stream>>>(
      bufB, wp_d1, wp_d2,
      wp_ih[0], wp_hh[0], wp_ih[1], wp_hh[1], wp_ih[2], wp_hh[2], wp_ow,
      d1_b, d2_b,
      g_bih[0], g_bhh[0], g_bih[1], g_bhh[1], g_bih[2], g_bhh[2],
      ow_b, (float*)d_out, steps, 200);
}
